// FNO2d_Multi_57389353009385
// MI455X (gfx1250) — matrix operand order repaired and checked by reading, not compiled, not run on hardware
//
#include <hip/hip_runtime.h>
#include <hip/hip_bf16.h>
#include <math.h>

// ---------------- problem constants ----------------
#define B_    32
#define V_    3
#define S_    106
#define T_IN_ 30
#define W_    32
#define NL_   6
#define SS_   (S_*S_)          // 11236
#define NPB_  (V_*SS_)         // 33708 points per batch
#define NTP_  ((NPB_ + 15) / 16) // 2107 point-tiles per batch
#define NSLAB_ (B_*W_*V_)      // 3072 (b,c,v) slabs
#define HN_   ((size_t)NSLAB_*SS_)

typedef _Float16 f16;
typedef __attribute__((ext_vector_type(16))) _Float16 v16h;
typedef __attribute__((ext_vector_type(8)))  float    v8f;

// ---------------- WMMA helper (gfx1250, wave32) ----------------
__device__ __forceinline__ v8f wmma16(v16h a, v16h b, v8f c) {
  return __builtin_amdgcn_wmma_f32_16x16x32_f16(false, a, false, b, (short)0, c,
                                                false, false);
}

// ---- fragment-major tile storage: tile = 32 lanes x 16 halfs (512 f16, 1KB) ----
// A tile (16x32, M x K): element (m,k) -> lane = m + 16*(k bit3), pos = (k>>4)*8 | (k&7)
__device__ __forceinline__ void stA(f16* base, int m, int k, float v) {
  int lane = m + ((k & 8) << 1);
  int pos  = ((k >> 4) << 3) | (k & 7);
  base[lane * 16 + pos] = (f16)v;
}
// B tile (32x16, K x N): element (k,n) -> lane = n + 16*(k bit3), pos = (k>>4)*8 | (k&7)
__device__ __forceinline__ void stB(f16* base, int k, int n, float v) {
  int lane = n + ((k & 8) << 1);
  base[lane * 16 + (((k >> 4) << 3) | (k & 7))] = (f16)v;
}
// load a lane's fragment: 32 contiguous bytes -> 2x b128 loads
__device__ __forceinline__ v16h ldFrag(const f16* base) {
  return *reinterpret_cast<const v16h*>(base + ((threadIdx.x & 31) << 4));
}

__device__ __forceinline__ float gelu_exact(float x) {
  return 0.5f * x * (1.0f + erff(x * 0.70710678118654752440f));
}

// ---------------- K0: build all constant operands in fragment layout ----------------
__global__ void k_tables(f16* gyF, f16* gxrF, f16* gxiF, f16* gxinF,
                         f16* exrF, f16* exiF, f16* exinF, f16* eyF,
                         const float* fc0_w, f16* fc0F,
                         const float* conv_w, f16* convF,
                         const float* fc1_w, f16* fc1F) {
  const float TP = 6.28318530717958647692f;
  for (int i = threadIdx.x; i < 128*32; i += blockDim.x) {
    int y = i >> 5, c = i & 31, ky = c & 15;
    float v = 0.f;
    if (y < S_) {
      float th = TP * (float)(ky * y) / (float)S_;
      v = (c < 16) ? __cosf(th) : -__sinf(th);          // e^{-i th}
    }
    stB(gyF + ((y >> 5) * 2 + (c >> 4)) * 512, y & 31, c & 15, v);
  }
  for (int i = threadIdx.x; i < 32*128; i += blockDim.x) {
    int kxi = i >> 7, xx = i & 127;
    float r = 0.f, im = 0.f;
    if (xx < S_) {
      int kx = (kxi < 16) ? kxi : kxi + 74;             // 90..105
      float th = TP * (float)(kx * xx) / (float)S_;
      r = __cosf(th); im = -__sinf(th);                 // e^{-i th}
    }
    int tile = ((kxi >> 4) * 4 + (xx >> 5)) * 512;
    int m = kxi & 15, k = xx & 31;
    stA(gxrF + tile, m, k, r);
    stA(gxiF + tile, m, k, im);
    stA(gxinF + tile, m, k, -im);
  }
  for (int i = threadIdx.x; i < 112*32; i += blockDim.x) {
    int xx = i >> 5, kxi = i & 31;
    float r = 0.f, im = 0.f;
    if (xx < S_) {
      int kx = (kxi < 16) ? kxi : kxi + 74;
      float th = TP * (float)(kx * xx) / (float)S_;
      r = __cosf(th); im = __sinf(th);                  // e^{+i th}
    }
    int tile = (xx >> 4) * 512;
    int m = xx & 15;
    stA(exrF + tile, m, kxi, r);
    stA(exiF + tile, m, kxi, im);
    stA(exinF + tile, m, kxi, -im);
  }
  for (int i = threadIdx.x; i < 32*112; i += blockDim.x) {
    int rr = i / 112, y = i % 112;
    float v = 0.f;
    if (y < S_) {
      int ky = rr & 15;
      float s = ((ky == 0) ? 1.f : 2.f) / (float)SS_;
      float th = TP * (float)(ky * y) / (float)S_;
      v = (rr < 16) ? s * __cosf(th) : -s * __sinf(th);
    }
    stB(eyF + (y >> 4) * 512, rr, y & 15, v);
  }
  for (int i = threadIdx.x; i < 32*32; i += blockDim.x) {     // fc0_w [t][c]
    int t = i >> 5, c = i & 31;
    stB(fc0F + (c >> 4) * 512, t, c & 15, fc0_w[i]);
  }
  for (int i = threadIdx.x; i < NL_*1024; i += blockDim.x) {  // conv_w [l][o][c]
    int l = i >> 10, rem = i & 1023, o = rem >> 5, c = rem & 31;
    stA(convF + (l * 2 + (o >> 4)) * 512, o & 15, c, conv_w[i]);
  }
  for (int i = threadIdx.x; i < 32*128; i += blockDim.x) {    // fc1_w [k][nc]
    int k = i >> 7, nc = i & 127;
    stB(fc1F + (nc >> 4) * 512, k, nc & 15, fc1_w[i]);
  }
  __builtin_amdgcn_s_wait_tensorcnt(0);
}

// ---------------- K1: lift (concat grid + fc0) -> h[b][c][v][x][y] ----------------
__global__ void k_fc0(const float* __restrict__ x, const float* __restrict__ gridx,
                      const float* __restrict__ gridy, const f16* __restrict__ fc0F,
                      const float* __restrict__ fc0_b, float* __restrict__ hout) {
  __shared__ alignas(32) f16 sA[4][512];
  int wave = threadIdx.x >> 5, lane = threadIdx.x & 31;
  int gw = blockIdx.x * 4 + wave;                    // bb*NTP_ + ntile
  int bb = gw / NTP_, ntile = gw % NTP_;
  int pbase = ntile * 16;                            // point-in-batch base
  bool interior = (pbase + 16 <= NPB_);
  f16* A = sA[wave];
  for (int i = lane; i < 512; i += 32) {
    int r = i >> 5, t = i & 31;
    int n0 = pbase + r;
    float v = 0.f;
    if (n0 < NPB_) {
      if (t < T_IN_) v = x[(bb * NPB_ + n0) * T_IN_ + t];
      else           v = (t == T_IN_) ? gridx[(n0 / S_) % S_] : gridy[n0 % S_];
    }
    stA(A, r, t, v);
  }
  __syncthreads();
  v16h a = ldFrag(A);
  int half8 = (lane >> 4) << 3;
#pragma unroll
  for (int nt = 0; nt < 2; ++nt) {
    v8f acc = {};
    acc = wmma16(a, ldFrag(fc0F + nt * 512), acc);
    int c = nt * 16 + (lane & 15);
    float bias = fc0_b[c];
    float* ob = hout + (bb * W_ + c) * NPB_ + pbase + half8;   // 16B aligned
    if (interior) {
      float4 lo, hi;
      lo.x = acc[0] + bias; lo.y = acc[1] + bias;
      lo.z = acc[2] + bias; lo.w = acc[3] + bias;
      hi.x = acc[4] + bias; hi.y = acc[5] + bias;
      hi.z = acc[6] + bias; hi.w = acc[7] + bias;
      *(float4*)(ob)     = lo;
      *(float4*)(ob + 4) = hi;
    } else {
#pragma unroll
      for (int j = 0; j < 8; ++j)
        if (pbase + half8 + j < NPB_) ob[j] = acc[j] + bias;
    }
  }
}

// ---------------- K2: forward DFT along y: t1[slab][x][ky | 16+ky] ----------------
__global__ void k_dfty(const float* __restrict__ h, const f16* __restrict__ gyF,
                       float* __restrict__ t1) {
  int wave = threadIdx.x >> 5, lane = threadIdx.x & 31;
  int gw = blockIdx.x * 4 + wave;                    // slab*7 + mt
  int slab = gw / 7, mt = gw % 7;
  const float* hb = h + slab * SS_;
  int m = lane & 15, kb8 = (lane >> 4) << 3;
  int xrow = mt * 16 + m;
  int rowoff = xrow * S_;
  v8f acc0 = {}, acc1 = {};
#pragma unroll
  for (int kc = 0; kc < 4; ++kc) {
    v16h a;
#pragma unroll
    for (int j = 0; j < 8; ++j) {
      int k = ((j & 3) << 1) + ((j >> 2) << 4) + kb8 + kc * 32;  // even
      float v0 = 0.f, v1 = 0.f;
      if (xrow < S_ && k < S_) {
        float2 p2 = *(const float2*)(hb + rowoff + k);
        v0 = p2.x; v1 = p2.y;
      }
      a[2*j] = (f16)v0; a[2*j+1] = (f16)v1;
    }
    acc0 = wmma16(a, ldFrag(gyF + (kc*2 + 0) * 512), acc0);      // real
    acc1 = wmma16(a, ldFrag(gyF + (kc*2 + 1) * 512), acc1);      // imag
  }
  float* tb = t1 + slab * (S_ * 32);
  int half8 = (lane >> 4) << 3, n = lane & 15;
#pragma unroll
  for (int j = 0; j < 8; ++j) {
    int xr = mt * 16 + half8 + j;
    if (xr < S_) {
      tb[xr * 32 + n]      = acc0[j];
      tb[xr * 32 + 16 + n] = acc1[j];
    }
  }
}

// ---------------- K2b: forward DFT along x: hf[slab][kx][ky | 16+ky] ----------------
__global__ void k_dftx(const float* __restrict__ t1, const f16* __restrict__ gxrF,
                       const f16* __restrict__ gxiF, const f16* __restrict__ gxinF,
                       float* __restrict__ hf) {
  int wave = threadIdx.x >> 5, lane = threadIdx.x & 31;
  int gw = blockIdx.x * 4 + wave;                    // slab*4 + (mt*2+half)
  int slab = gw >> 2, sub = gw & 3, mt = sub >> 1, half = sub & 1;
  const float* tb = t1 + slab * (S_ * 32);
  int n = lane & 15, kb16 = (lane >> 4) << 4;
  v8f acc = {};
#pragma unroll
  for (int kc = 0; kc < 4; ++kc) {
    v16h br, bi;
#pragma unroll
    for (int j = 0; j < 8; ++j) {
      int k0 = ((j & 3) << 1) + ((j >> 2) << 4) + (kb16 >> 1) + kc * 32; // even; k0<S_ => k0+1<S_
      float r0 = 0.f, r1 = 0.f, i0 = 0.f, i1 = 0.f;
      if (k0 < S_) {
        r0 = tb[k0*32 + n];       i0 = tb[k0*32 + 16 + n];
        r1 = tb[(k0+1)*32 + n];   i1 = tb[(k0+1)*32 + 16 + n];
      }
      br[2*j] = (f16)r0; br[2*j+1] = (f16)r1;
      bi[2*j] = (f16)i0; bi[2*j+1] = (f16)i1;
    }
    int tile = (mt * 4 + kc) * 512;
    v16h ar = ldFrag(gxrF + tile);
    if (half == 0) {                                 // Re = gxr*t1r - gxi*t1i
      acc = wmma16(ar, br, acc);
      acc = wmma16(ldFrag(gxinF + tile), bi, acc);
    } else {                                         // Im = gxr*t1i + gxi*t1r
      acc = wmma16(ar, bi, acc);
      acc = wmma16(ldFrag(gxiF + tile), br, acc);
    }
  }
  float* ob = hf + slab * 1024;
  int half8 = (lane >> 4) << 3;
#pragma unroll
  for (int j = 0; j < 8; ++j) {
    int kx = mt * 16 + half8 + j;
    ob[kx * 32 + half * 16 + n] = acc[j];
  }
}

// ---------------- K3: per-mode complex channel mixing (32b x 96 x 96) ----------------
__global__ void k_mix(const float* __restrict__ hf,
                      const float* __restrict__ w1r, const float* __restrict__ w1i,
                      const float* __restrict__ w2r, const float* __restrict__ w2i,
                      float* __restrict__ hf2, int layer) {
  __shared__ alignas(32) f16 ArF[6][512], AiF[6][512], AnF[6][512]; // [mt*3+kc]
  __shared__ alignas(32) f16 BrF[18][512], BiF[18][512];            // [kc*6+nt]
  int kxi = blockIdx.x >> 4, ky = blockIdx.x & 15;
  for (int i = threadIdx.x; i < 32*96; i += blockDim.x) {
    int b = i / 96, ip = i % 96;
    const float* base = hf + (b*96 + ip) * 1024 + kxi*32 + ky;
    float r = base[0], im = base[16];
    int mt = b >> 4, kc = ip >> 5, m = b & 15, k = ip & 31;
    stA(ArF[mt*3+kc], m, k, r);
    stA(AiF[mt*3+kc], m, k, im);
    stA(AnF[mt*3+kc], m, k, -im);
  }
  const float* wr = (kxi < 16) ? w1r : w2r;
  const float* wi = (kxi < 16) ? w1i : w2i;
  int moff = (kxi & 15) * 16 + ky;
  for (int i = threadIdx.x; i < 96*96; i += blockDim.x) {
    int ip = i / 96, jq = i % 96;
    int ii = ip / 3, p = ip % 3, jj = jq / 3, q = jq % 3;
    int idx = ((((layer*32 + ii)*32 + jj)*3 + p)*3 + q) * 256 + moff;
    int kc = ip >> 5, k = ip & 31, nt = jq >> 4, n = jq & 15;
    stB(BrF[kc*6+nt], k, n, wr[idx]);
    stB(BiF[kc*6+nt], k, n, wi[idx]);
  }
  __syncthreads();
  int wave = threadIdx.x >> 5, lane = threadIdx.x & 31;
  int half8 = (lane >> 4) << 3;
#pragma unroll
  for (int t = 0; t < 3; ++t) {
    int pair = wave * 3 + t;                         // 12 = 2 mtiles x 6 ntiles
    int mt = pair / 6, nt = pair % 6;
    v8f cre = {}, cim = {};
#pragma unroll
    for (int kc = 0; kc < 3; ++kc) {
      v16h fAr = ldFrag(ArF[mt*3+kc]);
      v16h fAi = ldFrag(AiF[mt*3+kc]);
      v16h fAn = ldFrag(AnF[mt*3+kc]);
      v16h fBr = ldFrag(BrF[kc*6+nt]);
      v16h fBi = ldFrag(BiF[kc*6+nt]);
      cre = wmma16(fAr, fBr, cre);
      cre = wmma16(fAn, fBi, cre);                   // - Ai*Bi
      cim = wmma16(fAr, fBi, cim);
      cim = wmma16(fAi, fBr, cim);
    }
#pragma unroll
    for (int j = 0; j < 8; ++j) {
      int b = mt * 16 + half8 + j;
      int jq = nt * 16 + (lane & 15);
      float* o = hf2 + (b*96 + jq) * 1024 + kxi*32 + ky;
      o[0] = cre[j]; o[16] = cim[j];
    }
  }
}

// ---------------- K4: inverse along x: t2[slab][x][ky | 16+ky] ----------------
__global__ void k_invx(const float* __restrict__ hf2, const f16* __restrict__ exrF,
                       const f16* __restrict__ exiF, const f16* __restrict__ exinF,
                       float* __restrict__ t2) {
  int wave = threadIdx.x >> 5, lane = threadIdx.x & 31;
  int gw = blockIdx.x * 4 + wave;                    // slab*7 + mt
  int slab = gw / 7, mt = gw % 7;
  const float* ob = hf2 + slab * 1024;
  int n = lane & 15, kb16 = (lane >> 4) << 4;
  v16h br, bi;
#pragma unroll
  for (int j = 0; j < 8; ++j) {
    int k0 = ((j & 3) << 1) + ((j >> 2) << 4) + (kb16 >> 1);
    br[2*j]   = (f16)ob[k0*32 + n];        br[2*j+1] = (f16)ob[(k0+1)*32 + n];
    bi[2*j]   = (f16)ob[k0*32 + 16 + n];   bi[2*j+1] = (f16)ob[(k0+1)*32 + 16 + n];
  }
  int tile = mt * 512;
  v8f cre = {}, cim = {};
  cre = wmma16(ldFrag(exrF + tile), br, cre);
  cre = wmma16(ldFrag(exinF + tile), bi, cre);
  cim = wmma16(ldFrag(exrF + tile), bi, cim);
  cim = wmma16(ldFrag(exiF + tile), br, cim);
  float* tb = t2 + slab * (S_ * 32);
  int half8 = (lane >> 4) << 3;
#pragma unroll
  for (int j = 0; j < 8; ++j) {
    int xr = mt * 16 + half8 + j;
    if (xr < S_) { tb[xr*32 + n] = cre[j]; tb[xr*32 + 16 + n] = cim[j]; }
  }
}

// ------- K5: inverse along y + add conv result + bias'd GELU (7 wmma / wave) -------
__global__ void k_invy(const float* __restrict__ t2, const f16* __restrict__ eyF,
                       float* __restrict__ hio, int do_gelu) {
  int wave = threadIdx.x >> 5, lane = threadIdx.x & 31;
  int gw = blockIdx.x * 4 + wave;                    // slab*7 + mt
  int slab = gw / 7, mt = gw % 7;
  const float* tb = t2 + slab * (S_ * 32);
  int m = lane & 15, kb8 = (lane >> 4) << 3;
  int xrow = mt * 16 + m;
  v16h a;
#pragma unroll
  for (int j = 0; j < 8; ++j) {
    int k = ((j & 3) << 1) + ((j >> 2) << 4) + kb8;  // even, < 32
    float v0 = 0.f, v1 = 0.f;
    if (xrow < S_) {
      float2 p2 = *(const float2*)(tb + xrow*32 + k);
      v0 = p2.x; v1 = p2.y;
    }
    a[2*j] = (f16)v0; a[2*j+1] = (f16)v1;
  }
  float* hb = hio + slab * SS_;
  int half8 = (lane >> 4) << 3, n = lane & 15;
#pragma unroll
  for (int nt = 0; nt < 7; ++nt) {
    v8f acc = {};
    acc = wmma16(a, ldFrag(eyF + nt * 512), acc);
    int yc = nt * 16 + n;
#pragma unroll
    for (int j = 0; j < 8; ++j) {
      int xr = mt * 16 + half8 + j;
      if (xr < S_ && yc < S_) {
        float v = acc[j] + hb[xr * S_ + yc];
        hb[xr * S_ + yc] = do_gelu ? gelu_exact(v) : v;
      }
    }
  }
}

// ---------------- K6: 1x1 conv over channels -> hout (with bias) ----------------
__global__ void k_conv(const float* __restrict__ hin, const f16* __restrict__ convF,
                       const float* __restrict__ conv_b, float* __restrict__ hout,
                       int layer) {
  int wave = threadIdx.x >> 5, lane = threadIdx.x & 31;
  int gw = blockIdx.x * 4 + wave;                    // b*NTP_ + ntile
  int b = gw / NTP_, ntile = gw % NTP_;
  bool interior = (ntile * 16 + 16 <= NPB_);
  const float* hb = hin + b * (W_ * NPB_);
  int n = lane & 15, kb16 = (lane >> 4) << 4;
  int col = ntile * 16 + n;
  v16h bf;
  if (interior) {
    const float* hc = hb + col;
    __builtin_prefetch(hc + kb16 * NPB_);
#pragma unroll
    for (int j = 0; j < 8; ++j) {
      int c0 = ((j & 3) << 1) + ((j >> 2) << 4) + (kb16 >> 1);
      bf[2*j]   = (f16)hc[c0 * NPB_];
      bf[2*j+1] = (f16)hc[(c0+1) * NPB_];
    }
  } else {
#pragma unroll
    for (int j = 0; j < 8; ++j) {
      int c0 = ((j & 3) << 1) + ((j >> 2) << 4) + (kb16 >> 1);
      float v0 = (col < NPB_) ? hb[c0 * NPB_ + col] : 0.f;
      float v1 = (col < NPB_) ? hb[(c0+1) * NPB_ + col] : 0.f;
      bf[2*j] = (f16)v0; bf[2*j+1] = (f16)v1;
    }
  }
  float* ob = hout + b * (W_ * NPB_);
  int half8 = (lane >> 4) << 3;
#pragma unroll
  for (int mt = 0; mt < 2; ++mt) {
    v8f acc = {};
    acc = wmma16(ldFrag(convF + (layer*2 + mt) * 512), bf, acc);
    float* oc = ob + (mt * 16 + half8) * NPB_ + col;
    if (interior) {
#pragma unroll
      for (int j = 0; j < 8; ++j)
        oc[j * NPB_] = acc[j] + conv_b[layer * 32 + mt * 16 + half8 + j];
    } else {
#pragma unroll
      for (int j = 0; j < 8; ++j)
        if (col < NPB_) oc[j * NPB_] = acc[j] + conv_b[layer * 32 + mt * 16 + half8 + j];
    }
  }
}

// ---------------- K7: fc1 + GELU + fc2 fused head ----------------
__global__ void k_head(const float* __restrict__ hin, const f16* __restrict__ fc1F,
                       const float* __restrict__ fc1_b, const float* __restrict__ fc2_w,
                       const float* __restrict__ fc2_b, float* __restrict__ out) {
  int wave = threadIdx.x >> 5, lane = threadIdx.x & 31;
  int gw = blockIdx.x * 4 + wave;                    // bb*NTP_ + ntile
  int bb = gw / NTP_, ntile = gw % NTP_;
  int pbase = ntile * 16;
  bool interior = (pbase + 16 <= NPB_);
  int m = lane & 15, kb8 = (lane >> 4) << 3;
  int n0 = pbase + m;
  v16h a;
  if (interior) {
    const float* hb = hin + bb * (W_ * NPB_) + n0;
#pragma unroll
    for (int j = 0; j < 8; ++j) {
      int k = ((j & 3) << 1) + ((j >> 2) << 4) + kb8;
      a[2*j]   = (f16)hb[k * NPB_];
      a[2*j+1] = (f16)hb[(k + 1) * NPB_];
    }
  } else {
    bool vrow = (n0 < NPB_);
    const float* hb = hin + bb * (W_ * NPB_) + (vrow ? n0 : 0);
#pragma unroll
    for (int j = 0; j < 8; ++j) {
      int k = ((j & 3) << 1) + ((j >> 2) << 4) + kb8;
      float v0 = vrow ? hb[k * NPB_] : 0.f;
      float v1 = vrow ? hb[(k + 1) * NPB_] : 0.f;
      a[2*j] = (f16)v0; a[2*j+1] = (f16)v1;
    }
  }
  float partial[8] = {0.f,0.f,0.f,0.f,0.f,0.f,0.f,0.f};
  int n = lane & 15;
#pragma unroll
  for (int nt = 0; nt < 8; ++nt) {
    int nc = nt * 16 + n;
    v8f acc = {};
    acc = wmma16(a, ldFrag(fc1F + nt * 512), acc);
    float bias = fc1_b[nc], w2 = fc2_w[nc];
#pragma unroll
    for (int j = 0; j < 8; ++j) partial[j] += gelu_exact(acc[j] + bias) * w2;
  }
  int half8 = (lane >> 4) << 3;
  float* ob = out + bb * NPB_ + pbase + half8;
#pragma unroll
  for (int j = 0; j < 8; ++j) {
    float v = partial[j];
    v += __shfl_xor(v, 1);
    v += __shfl_xor(v, 2);
    v += __shfl_xor(v, 4);
    v += __shfl_xor(v, 8);
    bool wr = interior || (pbase + half8 + j < NPB_);
    if ((lane & 15) == 0 && wr) ob[j] = v + fc2_b[0];
  }
}

// ---------------- launcher ----------------
extern "C" void kernel_launch(void* const* d_in, const int* in_sizes, int n_in,
                              void* d_out, int out_size, void* d_ws, size_t ws_size,
                              hipStream_t stream) {
  const float* x     = (const float*)d_in[0];
  const float* gridx = (const float*)d_in[1];
  const float* gridy = (const float*)d_in[2];
  const float* fc0_w = (const float*)d_in[3];
  const float* fc0_b = (const float*)d_in[4];
  const float* w1r   = (const float*)d_in[5];
  const float* w1i   = (const float*)d_in[6];
  const float* w2r   = (const float*)d_in[7];
  const float* w2i   = (const float*)d_in[8];
  const float* convw = (const float*)d_in[9];
  const float* convb = (const float*)d_in[10];
  const float* fc1_w = (const float*)d_in[11];
  const float* fc1_b = (const float*)d_in[12];
  const float* fc2_w = (const float*)d_in[13];
  const float* fc2_b = (const float*)d_in[14];
  (void)in_sizes; (void)n_in; (void)out_size; (void)ws_size;

  char* wsb = (char*)d_ws;
  size_t off = 0;
  auto take = [&](size_t bytes) -> void* {
    void* p = wsb + off;
    off = (off + bytes + 255) & ~(size_t)255;
    return p;
  };
  float* hA  = (float*)take(HN_ * sizeof(float));
  float* hB  = (float*)take(HN_ * sizeof(float));
  float* tb  = (float*)take((size_t)NSLAB_ * S_ * 32 * sizeof(float)); // t1/t2
  float* hf  = (float*)take((size_t)NSLAB_ * 1024 * sizeof(float));
  float* hf2 = (float*)take((size_t)NSLAB_ * 1024 * sizeof(float));
  f16* gyF   = (f16*)take(8  * 512 * sizeof(f16));
  f16* gxrF  = (f16*)take(8  * 512 * sizeof(f16));
  f16* gxiF  = (f16*)take(8  * 512 * sizeof(f16));
  f16* gxinF = (f16*)take(8  * 512 * sizeof(f16));
  f16* exrF  = (f16*)take(7  * 512 * sizeof(f16));
  f16* exiF  = (f16*)take(7  * 512 * sizeof(f16));
  f16* exinF = (f16*)take(7  * 512 * sizeof(f16));
  f16* eyF   = (f16*)take(7  * 512 * sizeof(f16));
  f16* fc0F  = (f16*)take(2  * 512 * sizeof(f16));
  f16* convF = (f16*)take(12 * 512 * sizeof(f16));
  f16* fc1F  = (f16*)take(8  * 512 * sizeof(f16));

  k_tables<<<1, 256, 0, stream>>>(gyF, gxrF, gxiF, gxinF, exrF, exiF, exinF, eyF,
                                  fc0_w, fc0F, convw, convF, fc1_w, fc1F);

  const int blk_pts  = (B_ * NTP_) / 4;               // 16856
  const int blk_7    = (NSLAB_ * 7) / 4;              // 5376
  const int blk_dftx = (NSLAB_ * 4) / 4;              // 3072

  k_fc0<<<blk_pts, 128, 0, stream>>>(x, gridx, gridy, fc0F, fc0_b, hA);

  float* cur = hA; float* nxt = hB;
  for (int l = 0; l < NL_; ++l) {
    k_conv<<<blk_pts,  128, 0, stream>>>(cur, convF, convb, nxt, l);
    k_dfty<<<blk_7,    128, 0, stream>>>(cur, gyF, tb);
    k_dftx<<<blk_dftx, 128, 0, stream>>>(tb, gxrF, gxiF, gxinF, hf);
    k_mix <<<512,      128, 0, stream>>>(hf, w1r, w1i, w2r, w2i, hf2, l);
    k_invx<<<blk_7,    128, 0, stream>>>(hf2, exrF, exiF, exinF, tb);
    k_invy<<<blk_7,    128, 0, stream>>>(tb, eyF, nxt, (l < 3) ? 1 : 0);
    float* t = cur; cur = nxt; nxt = t;
  }

  k_head<<<blk_pts, 128, 0, stream>>>(cur, fc1F, fc1_b, fc2_w, fc2_b,
                                      (float*)d_out);
}
